// CrossAttention_38517266711011
// MI455X (gfx1250) — matrix operand order repaired and checked by reading, not compiled, not run on hardware
//
#include <hip/hip_runtime.h>

// MI455X / gfx1250, wave32. bf16 WMMA GEMM path.
typedef __attribute__((ext_vector_type(16))) __bf16 v16bf;
typedef __attribute__((ext_vector_type(8)))  float  v8f;

#define D512 512

// f32 -> bf16 via hardware convert (compiler picks v_cvt_*bf16* on gfx1250,
// RNE). Much cheaper than the manual bit-twiddled rounding sequence.
__device__ __forceinline__ unsigned int cvt2bf(float a, float b) {
    unsigned short lo = __builtin_bit_cast(unsigned short, (__bf16)a);
    unsigned short hi = __builtin_bit_cast(unsigned short, (__bf16)b);
    return (unsigned int)lo | ((unsigned int)hi << 16);
}

__device__ __forceinline__ unsigned long long pack4(float4 v) {
    return (unsigned long long)cvt2bf(v.x, v.y)
         | ((unsigned long long)cvt2bf(v.z, v.w) << 32);
}

// 32-byte fragment load from LDS (2x b128), 16B-aligned addresses.
__device__ __forceinline__ v16bf load_frag16(const unsigned short* p) {
    struct U { uint4 a, b; } u;
    u.a = *reinterpret_cast<const uint4*>(p);
    u.b = *reinterpret_cast<const uint4*>(p + 8);
    return __builtin_bit_cast(v16bf, u);
}

// A-layout permutation within a 32-wide k chunk:
// k 0..15 -> pos 0..15 (lanes 0-15 half),
// k 16..31 -> pos 16..31 (lanes 16-31 half).
__device__ __forceinline__ int posbase(int g) {   // g = (k%32)>>3
    return (int)((0x18100800u >> (g << 3)) & 0xffu);
}

// ---------------------------------------------------------------------------
// Generic C[M,512] = X[M,512] @ W[512,512]^T + bias, M_BLK=128, N_BLK=64.
// grid = (M/128, 8), block = 256 (8 waves), each wave: 16 rows x 64 cols.
// ---------------------------------------------------------------------------
__global__ __launch_bounds__(256) void wmma_gemm_bias_kernel(
    const float* __restrict__ X, const float* __restrict__ W,
    const float* __restrict__ bias, float* __restrict__ C)
{
    __shared__ unsigned short Als[128 * 40];   // 10 KB, padded stride 40 us (80 B)
    __shared__ unsigned short Bls[64 * 40];    //  5 KB

    const int t    = threadIdx.x;
    const int w    = t >> 5;
    const int lane = t & 31;
    const int l15  = lane & 15;
    const int hi   = (lane & 16) ? 16 : 0;
    const int rowb = blockIdx.x * 128;
    const int colb = blockIdx.y * 64;

    v8f acc[4];
#pragma unroll
    for (int nt = 0; nt < 4; ++nt) acc[nt] = (v8f)0.0f;

    for (int kc = 0; kc < 16; ++kc) {
        __syncthreads();
        // Stage A tile 128x32 f32 -> bf16 permuted (1024 float4 / 256 thr)
#pragma unroll
        for (int q = 0; q < 4; ++q) {
            int f4 = t + q * 256;
            int r  = f4 >> 3;
            int kq = (f4 & 7) << 2;
            float4 v = *(const float4*)(X + (size_t)(rowb + r) * D512 + kc * 32 + kq);
            *(unsigned long long*)(Als + r * 40 + posbase(kq >> 3) + (kq & 7)) = pack4(v);
        }
        // Stage B tile 64x32 f32 -> bf16 contiguous (512 float4 / 256 thr)
#pragma unroll
        for (int q = 0; q < 2; ++q) {
            int f4 = t + q * 256;
            int n  = f4 >> 3;
            int kq = (f4 & 7) << 2;
            float4 v = *(const float4*)(W + (size_t)(colb + n) * D512 + kc * 32 + kq);
            *(unsigned long long*)(Bls + n * 40 + kq) = pack4(v);
        }
        __syncthreads();

        v16bf a = load_frag16(Als + (w * 16 + l15) * 40 + hi);
#pragma unroll
        for (int nt = 0; nt < 4; ++nt) {
            v16bf b = load_frag16(Bls + (nt * 16 + l15) * 40 + hi);
            acc[nt] = __builtin_amdgcn_wmma_f32_16x16x32_bf16(
                false, a, false, b, (short)0, acc[nt], false, false);
        }
    }

    const int mrow = rowb + w * 16 + ((lane & 16) ? 8 : 0);
#pragma unroll
    for (int nt = 0; nt < 4; ++nt) {
        int col  = colb + nt * 16 + l15;
        float bb = bias[col];
#pragma unroll
        for (int r = 0; r < 8; ++r)
            C[(size_t)(mrow + r) * D512 + col] = acc[nt][r] + bb;
    }
}

// ---------------------------------------------------------------------------
// Fused K/V projection + local attention. One WG = 4 anchors (128 neighbor
// rows). Neighbor tile staged once in LDS as bf16; per head: 128x64 K and V
// projections via WMMA, then wave-per-anchor softmax over 32 neighbors.
// Dynamic LDS = 220160 B.
// ---------------------------------------------------------------------------
#define XSTR 528   // 512 + 16 us pad  (1056 B row, 16B/32B aligned)
#define WSTR 40    // 32 + 8 us pad
#define KVSTR 65   // 64 + 1 f32 pad

__global__ __launch_bounds__(256) void kv_attn_kernel(
    const float* __restrict__ Nb,   // [131072, 512] neighbor tokens
    const float* __restrict__ Wk, const float* __restrict__ bk,
    const float* __restrict__ Wv, const float* __restrict__ bv,
    const float* __restrict__ Qws,  // [4096, 512] projected Q
    float* __restrict__ Ao)         // [4096, 512] attention output
{
    extern __shared__ unsigned char smem[];
    unsigned short* Xls = (unsigned short*)smem;                          // 128*528 us
    float*          Qls = (float*)(smem + 128 * XSTR * 2);                // 4*512 f32
    unsigned short* Wls = (unsigned short*)(smem + 128 * XSTR * 2 + 8192);// 2*64*40 us
    float*          Kls = (float*)(smem + 128 * XSTR * 2 + 8192 + 10240); // 128*65 f32
    float*          Vls = Kls + 128 * KVSTR;                              // 128*65 f32

    const int t    = threadIdx.x;
    const int w    = t >> 5;
    const int lane = t & 31;
    const int l15  = lane & 15;
    const int hi   = (lane & 16) ? 16 : 0;
    const int rowb = blockIdx.x * 128;   // neighbor row base
    const int ab   = blockIdx.x * 4;     // anchor base

    // Stage neighbor tile 128x512 f32 -> bf16 (A-permuted). 16384 f4 / 256 thr.
    for (int q = 0; q < 64; ++q) {
        int f4 = t + q * 256;
        int r  = f4 >> 7;
        int kq = (f4 & 127) << 2;
        float4 v = *(const float4*)(Nb + (size_t)(rowb + r) * D512 + kq);
        int c  = kq >> 5;
        int kk = kq & 31;
        *(unsigned long long*)(Xls + r * XSTR + c * 32 + posbase(kk >> 3) + (kk & 7)) = pack4(v);
    }
    // Stage Q rows for the 4 anchors (f32).
#pragma unroll
    for (int q = 0; q < 2; ++q) {
        int f4 = t + q * 256;
        *(float4*)(Qls + f4 * 4) = *(const float4*)(Qws + (size_t)ab * D512 + f4 * 4);
    }
    __syncthreads();

    for (int h = 0; h < 8; ++h) {
        v8f accK[4], accV[4];
#pragma unroll
        for (int nt = 0; nt < 4; ++nt) { accK[nt] = (v8f)0.0f; accV[nt] = (v8f)0.0f; }

        for (int kc = 0; kc < 16; ++kc) {
            __syncthreads();
            // Stage Wk/Wv head-h chunk: 2 x 64x32 f32 -> bf16 (1024 f4 / 256 thr)
#pragma unroll
            for (int q = 0; q < 4; ++q) {
                int f4  = t + q * 256;
                int mat = f4 >> 9;
                int idx = f4 & 511;
                int n   = idx >> 3;
                int kq  = (idx & 7) << 2;
                const float* Ws = mat ? Wv : Wk;
                float4 v = *(const float4*)(Ws + (size_t)(h * 64 + n) * D512 + kc * 32 + kq);
                *(unsigned long long*)(Wls + mat * (64 * WSTR) + n * WSTR + kq) = pack4(v);
            }
            __syncthreads();

            v16bf a = load_frag16(Xls + (w * 16 + l15) * XSTR + kc * 32 + hi);
#pragma unroll
            for (int nt = 0; nt < 4; ++nt) {
                v16bf b0 = load_frag16(Wls + (nt * 16 + l15) * WSTR + hi);
                accK[nt] = __builtin_amdgcn_wmma_f32_16x16x32_bf16(
                    false, a, false, b0, (short)0, accK[nt], false, false);
                v16bf b1 = load_frag16(Wls + 64 * WSTR + (nt * 16 + l15) * WSTR + hi);
                accV[nt] = __builtin_amdgcn_wmma_f32_16x16x32_bf16(
                    false, a, false, b1, (short)0, accV[nt], false, false);
            }
        }
        __syncthreads();

        // D-tile -> LDS (with bias): lane<16 rows r, lane>=16 rows r+8.
        const int rloc = w * 16 + ((lane & 16) ? 8 : 0);
#pragma unroll
        for (int nt = 0; nt < 4; ++nt) {
            int col  = nt * 16 + l15;
            float bK = bk[h * 64 + col];
            float bV = bv[h * 64 + col];
#pragma unroll
            for (int r = 0; r < 8; ++r) {
                Kls[(rloc + r) * KVSTR + col] = accK[nt][r] + bK;
                Vls[(rloc + r) * KVSTR + col] = accV[nt][r] + bV;
            }
        }
        __syncthreads();

        // Attention: waves 0..3 -> anchor w; lane j = neighbor j (wave32).
        if (w < 4) {
            const float* q    = Qls + w * D512 + h * 64;
            const float* krow = Kls + (w * 32 + lane) * KVSTR;
            float s = 0.0f;
#pragma unroll
            for (int d = 0; d < 64; ++d) s += q[d] * krow[d];
            s *= 0.125f;                               // 1/sqrt(64)
            float m = s;
#pragma unroll
            for (int off = 16; off > 0; off >>= 1) m = fmaxf(m, __shfl_xor(m, off));
            float e = __expf(s - m);
            float sum = e;
#pragma unroll
            for (int off = 16; off > 0; off >>= 1) sum += __shfl_xor(sum, off);
            float wt = e / sum;

            float a0 = 0.0f, a1 = 0.0f;
#pragma unroll
            for (int j = 0; j < 32; ++j) {
                float wj = __shfl(wt, j);
                const float* vrow = Vls + (w * 32 + j) * KVSTR;
                a0 += wj * vrow[lane];
                a1 += wj * vrow[lane + 32];
            }
            float* o = Ao + (size_t)(ab + w) * D512 + h * 64;
            o[lane]      = a0;
            o[lane + 32] = a1;
        }
        // Next head's first __syncthreads() in the kc loop fences Kls/Vls reuse.
    }
}

extern "C" void kernel_launch(void* const* d_in, const int* in_sizes, int n_in,
                              void* d_out, int out_size, void* d_ws, size_t ws_size,
                              hipStream_t stream) {
    (void)in_sizes; (void)n_in; (void)out_size; (void)ws_size;
    const float* anchor   = (const float*)d_in[0];  // [2,2048,512]
    const float* neighbor = (const float*)d_in[1];  // [2,2048,32,512]
    const float* Wq = (const float*)d_in[2];
    const float* bq = (const float*)d_in[3];
    const float* Wk = (const float*)d_in[4];
    const float* bk = (const float*)d_in[5];
    const float* Wv = (const float*)d_in[6];
    const float* bv = (const float*)d_in[7];
    const float* Wo = (const float*)d_in[8];
    const float* bo = (const float*)d_in[9];

    float* Qws  = (float*)d_ws;                     // 4096*512 f32 = 8 MB
    float* Aows = Qws + (size_t)4096 * 512;         // 4096*512 f32 = 8 MB

    // 1) Q projection
    wmma_gemm_bias_kernel<<<dim3(4096 / 128, 8), 256, 0, stream>>>(anchor, Wq, bq, Qws);

    // 2) Fused K/V projection + local attention
    const size_t SM2 = (size_t)128 * XSTR * 2 + 8192 + 10240 + (size_t)2 * 128 * KVSTR * 4;
    hipFuncSetAttribute((const void*)kv_attn_kernel,
                        hipFuncAttributeMaxDynamicSharedMemorySize, (int)SM2);
    kv_attn_kernel<<<131072 / 128, 256, SM2, stream>>>(neighbor, Wk, bk, Wv, bv, Qws, Aows);

    // 3) Output projection
    wmma_gemm_bias_kernel<<<dim3(4096 / 128, 8), 256, 0, stream>>>(Aows, Wo, bo, (float*)d_out);
}
